// Pointnet2Backbone_11132555231470
// MI455X (gfx1250) — matrix operand order repaired and checked by reading, not compiled, not run on hardware
//
#include <hip/hip_runtime.h>
#include <hip/hip_bf16.h>
#include <cstddef>

// ---------------------------------------------------------------------------
// CDNA5 (gfx1250) PointNet++ backbone.
// Shared MLPs run on v_wmma_f32_16x16x32_bf16 (wave32). A/B operands are
// pre-packed into WMMA fragment layout (bf16, K zero-padded to 32) by small
// conversion kernels so the GEMM inner loop is pure b128 loads + wmma.
// ---------------------------------------------------------------------------

typedef __attribute__((ext_vector_type(16))) __bf16 v16bf;
typedef __attribute__((ext_vector_type(8)))  float  v8f;

#define WAVE 32

// -------------------------- Farthest point sampling ------------------------
__global__ __launch_bounds__(256) void fps_kernel(
    const float* __restrict__ xyz, int N, int npoint, int* __restrict__ fidx)
{
    __shared__ float dists[8192];
    __shared__ float redv[256];
    __shared__ int   redi[256];
    __shared__ int   cur;

    const int b   = blockIdx.x;
    const int tid = threadIdx.x;
    const int nt  = blockDim.x;
    const float* p = xyz + (size_t)b * N * 3;

    for (int i = tid; i < N; i += nt) dists[i] = 1e10f;
    if (tid == 0) cur = 0;
    __syncthreads();

    for (int it = 0; it < npoint; ++it) {
        const int c = cur;
        if (tid == 0) fidx[b * npoint + it] = c;
        const float cx = p[c * 3 + 0], cy = p[c * 3 + 1], cz = p[c * 3 + 2];
        float best = -1.0f; int bi = 0;
        for (int i = tid; i < N; i += nt) {
            const float dx = p[i * 3 + 0] - cx;
            const float dy = p[i * 3 + 1] - cy;
            const float dz = p[i * 3 + 2] - cz;
            float d = dx * dx + dy * dy + dz * dz;
            d = fminf(dists[i], d);
            dists[i] = d;
            if (d > best) { best = d; bi = i; }
        }
        redv[tid] = best; redi[tid] = bi;
        __syncthreads();
        for (int s = nt >> 1; s > 0; s >>= 1) {
            if (tid < s && redv[tid + s] > redv[tid]) {
                redv[tid] = redv[tid + s]; redi[tid] = redi[tid + s];
            }
            __syncthreads();
        }
        if (tid == 0) cur = redi[0];
        __syncthreads();
    }
}

// ------------------------------ Gather sampled xyz -------------------------
__global__ void gather_xyz_kernel(const float* __restrict__ xyz,
                                  const int* __restrict__ fidx,
                                  int B, int N, int M, float* __restrict__ out)
{
    const int t = blockIdx.x * blockDim.x + threadIdx.x;
    if (t >= B * M) return;
    const int b = t / M;
    const int i = fidx[t];
    const float* s = xyz + ((size_t)b * N + i) * 3;
    out[t * 3 + 0] = s[0];
    out[t * 3 + 1] = s[1];
    out[t * 3 + 2] = s[2];
}

// --------------------------------- Ball query ------------------------------
__global__ void ball_query_kernel(const float* __restrict__ xyz,
                                  const float* __restrict__ nxyz,
                                  int B, int N, int M, float r2,
                                  int* __restrict__ gidx)
{
    const int t = blockIdx.x * blockDim.x + threadIdx.x;
    if (t >= B * M) return;
    const int b = t / M;
    const float* q    = nxyz + (size_t)t * 3;
    const float* base = xyz + (size_t)b * N * 3;
    int* out = gidx + (size_t)t * 32;
    const float qx = q[0], qy = q[1], qz = q[2];
    int cnt = 0, first = 0;
    for (int i = 0; i < N && cnt < 32; ++i) {
        const float dx = base[3 * i + 0] - qx;
        const float dy = base[3 * i + 1] - qy;
        const float dz = base[3 * i + 2] - qz;
        if (dx * dx + dy * dy + dz * dz < r2) {
            if (cnt == 0) first = i;
            out[cnt++] = i;
        }
    }
    for (int k = cnt; k < 32; ++k) out[k] = first;
}

// ------------------------- Grouping (build GEMM input X) -------------------
// X is [3+Cprev][B*M*32] channel-major, column p = (b*M+j)*32 + s.
__global__ void group_kernel(const float* __restrict__ xyz,
                             const float* __restrict__ nxyz,
                             const float* __restrict__ feats,
                             const int* __restrict__ gidx,
                             int B, int N, int M, int Cprev,
                             float* __restrict__ Xo)
{
    const size_t P = (size_t)B * M * 32;
    const size_t t = (size_t)blockIdx.x * blockDim.x + threadIdx.x;
    if (t >= P) return;
    const int bj = (int)(t >> 5);
    const int b  = bj / M;
    const int i  = gidx[t];
    const float* pj = nxyz + (size_t)bj * 3;
    const float* pi = xyz + ((size_t)b * N + i) * 3;
    Xo[t]          = pi[0] - pj[0];
    Xo[P + t]      = pi[1] - pj[1];
    Xo[2 * P + t]  = pi[2] - pj[2];
    for (int c = 0; c < Cprev; ++c)
        Xo[(size_t)(3 + c) * P + t] =
            feats[(size_t)c * B * N + (size_t)b * N + i];
}

// ---------------- Pack W into WMMA A-fragment layout (bf16) ----------------
// Afrag[mtile][kstep][lane][16]; A 16x32 bf16 ISA layout:
//   lanes 0-15: row M = lane,    K = {0..7, 16..23}
//   lanes16-31: row M = lane-16, K = {8..15, 24..31}
__global__ void convA_kernel(const float* __restrict__ W,
                             __bf16* __restrict__ Af,
                             int M, int K, int K32)
{
    const int ks = K32 >> 5;
    const size_t tot = (size_t)(M >> 4) * ks * 512;
    const size_t t = (size_t)blockIdx.x * blockDim.x + threadIdx.x;
    if (t >= tot) return;
    const int j    = (int)(t & 15);
    const int lane = (int)((t >> 4) & 31);
    const size_t blk = t >> 9;
    const int mt = (int)(blk / ks);
    const int s  = (int)(blk % ks);
    const int mrow  = mt * 16 + (lane & 15);
    const int khalf = (lane >> 4) << 3;
    const int kk = (j < 8) ? (khalf + j) : (16 + khalf + (j - 8));
    const int k = s * 32 + kk;
    const float v = (k < K) ? W[(size_t)mrow * K + k] : 0.0f;
    Af[t] = (__bf16)v;
}

// ---------------- Pack X into WMMA B-fragment layout (bf16) ----------------
// Bfrag[ntile][kstep][lane][16]; B 32x16 bf16 ISA layout:
//   N = lane&15; lanes 0-15 hold K {0..7, 16..23}, lanes 16-31 hold K {8..15, 24..31}.
__global__ void convB_kernel(const float* __restrict__ X,
                             __bf16* __restrict__ Bf,
                             int K, int N, int K32)
{
    const int ks = K32 >> 5;
    const size_t tot = (size_t)(N >> 4) * ks * 512;
    const size_t t = (size_t)blockIdx.x * blockDim.x + threadIdx.x;
    if (t >= tot) return;
    const int j    = (int)(t & 15);
    const int lane = (int)((t >> 4) & 31);
    const size_t blk = t >> 9;
    const int nt = (int)(blk / ks);
    const int s  = (int)(blk % ks);
    const int n = nt * 16 + (lane & 15);
    const int khalf = (lane >> 4) << 3;
    const int k = s * 32 + ((j < 8) ? (khalf + j) : (16 + khalf + (j - 8)));
    const float v = (k < K) ? X[(size_t)k * N + n] : 0.0f;
    Bf[t] = (__bf16)v;
}

// ------------------- Fused GEMM + BatchNorm + ReLU via WMMA ----------------
// Y[M,N] = relu(bn(W*X)). One wave computes a 16x64 tile: one A fragment
// reused across 4 B fragments -> 4 wmma per k-step, pure b128 loads.
__global__ __launch_bounds__(32) void gemm_frag_wmma(
    const __bf16* __restrict__ Af, const __bf16* __restrict__ Bf,
    float* __restrict__ Y,
    const float* __restrict__ gam, const float* __restrict__ bet,
    const float* __restrict__ mu,  const float* __restrict__ var,
    int M, int K32, int N)
{
    const int lane = threadIdx.x;
    const int mt   = blockIdx.y;
    const int nt0  = blockIdx.x << 2;      // 4 consecutive n-tiles
    const int ks   = K32 >> 5;

    const v16bf* Ap = (const v16bf*)Af + (size_t)mt  * ks * 32 + lane;
    const v16bf* Bp = (const v16bf*)Bf + (size_t)nt0 * ks * 32 + lane;

    v8f acc0 = {}, acc1 = {}, acc2 = {}, acc3 = {};

    for (int s = 0; s < ks; ++s) {
        const v16bf a  = Ap[(size_t)s * 32];
        const v16bf b0 = Bp[(size_t)s * 32];
        const v16bf b1 = Bp[(size_t)(ks + s) * 32];
        const v16bf b2 = Bp[(size_t)(2 * ks + s) * 32];
        const v16bf b3 = Bp[(size_t)(3 * ks + s) * 32];
        acc0 = __builtin_amdgcn_wmma_f32_16x16x32_bf16(
            false, a, false, b0, (short)0, acc0, false, false);
        acc1 = __builtin_amdgcn_wmma_f32_16x16x32_bf16(
            false, a, false, b1, (short)0, acc1, false, false);
        acc2 = __builtin_amdgcn_wmma_f32_16x16x32_bf16(
            false, a, false, b2, (short)0, acc2, false, false);
        acc3 = __builtin_amdgcn_wmma_f32_16x16x32_bf16(
            false, a, false, b3, (short)0, acc3, false, false);
    }

    // C/D layout: VGPR r -> M = r + (lane>>4)*8, N = lane&15.
    const int mhalf = (lane >> 4) << 3;
    const int ncol  = lane & 15;
#pragma unroll
    for (int r = 0; r < 8; ++r) {
        const int m = mt * 16 + mhalf + r;
        const float sc = gam[m] * rsqrtf(var[m] + 1e-5f);
        const float sh = bet[m] - mu[m] * sc;
        float* yrow = Y + (size_t)m * N + (size_t)nt0 * 16 + ncol;
        yrow[0]  = fmaxf(acc0[r] * sc + sh, 0.0f);
        yrow[16] = fmaxf(acc1[r] * sc + sh, 0.0f);
        yrow[32] = fmaxf(acc2[r] * sc + sh, 0.0f);
        yrow[48] = fmaxf(acc3[r] * sc + sh, 0.0f);
    }
}

// ----------------------------- Max over nsample ----------------------------
__global__ void maxpool_kernel(const float* __restrict__ Y,
                               int B, int M, int C, float* __restrict__ out)
{
    const size_t tot = (size_t)C * B * M;
    const size_t t = (size_t)blockIdx.x * blockDim.x + threadIdx.x;
    if (t >= tot) return;
    const int c = (int)(t / ((size_t)B * M));
    const int q = (int)(t % ((size_t)B * M));
    const float* src = Y + ((size_t)c * B * M + q) * 32;
    float m = src[0];
#pragma unroll
    for (int s = 1; s < 32; ++s) m = fmaxf(m, src[s]);
    out[t] = m;
}

// ------------------------------ 3-NN + weights -----------------------------
__global__ void three_nn_kernel(const float* __restrict__ unk,
                                const float* __restrict__ kn,
                                int B, int Nu, int Nk,
                                int* __restrict__ idx3, float* __restrict__ w3)
{
    const int t = blockIdx.x * blockDim.x + threadIdx.x;
    if (t >= B * Nu) return;
    const int b = t / Nu;
    const float* q = unk + (size_t)t * 3;
    const float ux = q[0], uy = q[1], uz = q[2];
    const float* base = kn + (size_t)b * Nk * 3;
    float d0 = 1e30f, d1 = 1e30f, d2 = 1e30f;
    int i0 = 0, i1 = 0, i2 = 0;
    for (int i = 0; i < Nk; ++i) {
        const float dx = base[3 * i + 0] - ux;
        const float dy = base[3 * i + 1] - uy;
        const float dz = base[3 * i + 2] - uz;
        const float d = dx * dx + dy * dy + dz * dz;
        if (d < d0)      { d2 = d1; i2 = i1; d1 = d0; i1 = i0; d0 = d; i0 = i; }
        else if (d < d1) { d2 = d1; i2 = i1; d1 = d;  i1 = i; }
        else if (d < d2) { d2 = d;  i2 = i; }
    }
    float w0 = 1.0f / (d0 + 1e-8f);
    float w1 = 1.0f / (d1 + 1e-8f);
    float w2 = 1.0f / (d2 + 1e-8f);
    const float s = w0 + w1 + w2;
    w0 /= s; w1 /= s; w2 /= s;
    idx3[t * 3 + 0] = i0; idx3[t * 3 + 1] = i1; idx3[t * 3 + 2] = i2;
    w3[t * 3 + 0] = w0;  w3[t * 3 + 1] = w1;  w3[t * 3 + 2] = w2;
}

// -------------------- 3-NN interpolation into GEMM input -------------------
__global__ void interp_kernel(const float* __restrict__ kf,
                              const int* __restrict__ idx3,
                              const float* __restrict__ w3,
                              int B, int Nu, int Nk, int C,
                              float* __restrict__ Xo, size_t P)
{
    const size_t tot = (size_t)C * P;
    const size_t t = (size_t)blockIdx.x * blockDim.x + threadIdx.x;
    if (t >= tot) return;
    const int c = (int)(t / P);
    const int q = (int)(t % P);
    const int b = q / Nu;
    const float* row = kf + (size_t)c * B * Nk + (size_t)b * Nk;
    const int*   id  = idx3 + (size_t)q * 3;
    const float* w   = w3   + (size_t)q * 3;
    Xo[t] = row[id[0]] * w[0] + row[id[1]] * w[1] + row[id[2]] * w[2];
}

// -------------------- Copy skip-features into GEMM input rows --------------
__global__ void copyrows_kernel(const float* __restrict__ src,
                                float* __restrict__ Xo, int rowoff, int C, size_t P)
{
    const size_t tot = (size_t)C * P;
    const size_t t = (size_t)blockIdx.x * blockDim.x + threadIdx.x;
    if (t >= tot) return;
    Xo[(size_t)rowoff * P + t] = src[t];
}

// ------------------- Final layout change: [C][B*N] -> (B,C,N) --------------
__global__ void final_transpose_kernel(const float* __restrict__ F,
                                       float* __restrict__ out,
                                       int B, int C, int N)
{
    const size_t tot = (size_t)B * C * N;
    const size_t t = (size_t)blockIdx.x * blockDim.x + threadIdx.x;
    if (t >= tot) return;
    const int n = (int)(t % N);
    const int c = (int)((t / N) % C);
    const int b = (int)(t / ((size_t)N * C));
    out[t] = F[(size_t)c * B * N + (size_t)b * N + n];
}

// ===========================================================================
// Host orchestration
// ===========================================================================
namespace {

struct Layer {
    const float *W, *g, *b, *m, *v;
    int cin, cout;
};

static const int SA_CHANS[4][4] = {{3, 32, 32, 64},
                                   {67, 64, 64, 128},
                                   {131, 128, 128, 256},
                                   {259, 256, 256, 512}};
static const int FP_CHANS[4][4] = {{128, 128, 128, 128},
                                   {320, 256, 128, 0},
                                   {384, 256, 256, 0},
                                   {768, 256, 256, 0}};
static const int FP_NL[4] = {3, 2, 2, 2};

inline dim3 g1(size_t n, int bs = 256) { return dim3((unsigned)((n + bs - 1) / bs)); }

} // namespace

extern "C" void kernel_launch(void* const* d_in, const int* in_sizes, int n_in,
                              void* d_out, int out_size, void* d_ws, size_t ws_size,
                              hipStream_t stream)
{
    const int B = 4;
    const int Ns[5] = {8192, 2048, 512, 128, 32};
    const float R2[4] = {0.01f, 0.04f, 0.16f, 0.64f};
    const int Cf[5] = {0, 64, 128, 256, 512};

    // ---- locate pointcloud and params in the flattened input list ----
    const float* pc;
    int p0;
    if (in_sizes[0] == B * 8192 * 3) { pc = (const float*)d_in[0]; p0 = 1; }
    else                             { pc = (const float*)d_in[n_in - 1]; p0 = 0; }
    const bool sa_first = (in_sizes[p0] == 32 * 3);   // SA1 layer0 W = 96

    Layer sa[4][3], fp[4][3];
    int cursor = p0;
    auto read_mlp = [&](Layer* Ls, const int* ch, int nl) {
        for (int l = 0; l < nl; ++l) {
            Ls[l].cin = ch[l];
            Ls[l].cout = ch[l + 1];
            Ls[l].W = (const float*)d_in[cursor++];
            Ls[l].g = (const float*)d_in[cursor++];
            Ls[l].b = (const float*)d_in[cursor++];
            Ls[l].m = (const float*)d_in[cursor++];
            Ls[l].v = (const float*)d_in[cursor++];
        }
    };
    if (sa_first) {
        for (int i = 0; i < 4; ++i) read_mlp(sa[i], SA_CHANS[i], 3);
        for (int i = 0; i < 4; ++i) read_mlp(fp[i], FP_CHANS[i], FP_NL[i]);
    } else {
        for (int i = 0; i < 4; ++i) read_mlp(fp[i], FP_CHANS[i], FP_NL[i]);
        for (int i = 0; i < 4; ++i) read_mlp(sa[i], SA_CHANS[i], 3);
    }

    // ---- workspace carve ----
    char* ws = (char*)d_ws;
    size_t off = 0;
    auto alloc = [&](size_t bytes) -> void* {
        off = (off + 255) & ~(size_t)255;
        void* p = ws + off;
        off += bytes;
        return p;
    };

    int*   fidx = (int*)alloc((size_t)B * 2048 * sizeof(int));
    int*   gidx = (int*)alloc((size_t)B * 2048 * 32 * sizeof(int));
    int*   idx3 = (int*)alloc((size_t)B * 8192 * 3 * sizeof(int));
    float* w3   = (float*)alloc((size_t)B * 8192 * 3 * sizeof(float));

    float* xyzl[5];
    xyzl[0] = nullptr; // pc
    for (int k = 1; k <= 4; ++k)
        xyzl[k] = (float*)alloc((size_t)B * Ns[k] * 3 * sizeof(float));

    float* featl[5];
    featl[0] = nullptr;
    for (int k = 1; k <= 4; ++k)
        featl[k] = (float*)alloc((size_t)Cf[k] * B * Ns[k] * sizeof(float));

    float* fpout[4];
    const int fpOutC[4] = {128, 128, 256, 256};
    for (int k = 0; k < 4; ++k)
        fpout[k] = (float*)alloc((size_t)fpOutC[k] * B * Ns[k] * sizeof(float));

    // WMMA fragment buffers (bf16)
    __bf16* AFRAG = (__bf16*)alloc((size_t)1  * 1024 * 1024);           // 1 MB
    __bf16* BFRAG = (__bf16*)alloc((size_t)18 * 1024 * 1024);           // 18 MB

    const size_t PINGSZ = (size_t)64 * B * 2048 * 32 * sizeof(float);   // 67 MB
    float* PING = (float*)alloc(PINGSZ);
    float* PONG = (float*)alloc(PINGSZ);
    (void)ws_size;

    // One MLP layer: pack fragments, then fused WMMA GEMM + BN + ReLU.
    auto run_layer = [&](const Layer& L, const float* Xsrc, float* Ydst,
                         size_t P) {
        const int K32 = (L.cin + 31) & ~31;
        const size_t ae = (size_t)(L.cout >> 4) * (K32 >> 5) * 512;
        convA_kernel<<<g1(ae), 256, 0, stream>>>(L.W, AFRAG, L.cout, L.cin, K32);
        const size_t be = (size_t)(P >> 4) * (K32 >> 5) * 512;
        convB_kernel<<<g1(be), 256, 0, stream>>>(Xsrc, BFRAG, L.cin, (int)P, K32);
        dim3 grid((unsigned)(P >> 6), (unsigned)(L.cout >> 4));
        gemm_frag_wmma<<<grid, WAVE, 0, stream>>>(
            AFRAG, BFRAG, Ydst, L.g, L.b, L.m, L.v, L.cout, K32, (int)P);
    };

    // =========================== SA modules ===============================
    for (int i = 0; i < 4; ++i) {
        const int N = Ns[i], M = Ns[i + 1];
        const float* xsrc = (i == 0) ? pc : xyzl[i];

        fps_kernel<<<B, 256, 0, stream>>>(xsrc, N, M, fidx);
        gather_xyz_kernel<<<g1((size_t)B * M), 256, 0, stream>>>(
            xsrc, fidx, B, N, M, xyzl[i + 1]);
        ball_query_kernel<<<g1((size_t)B * M), 256, 0, stream>>>(
            xsrc, xyzl[i + 1], B, N, M, R2[i], gidx);

        const size_t P = (size_t)B * M * 32;
        group_kernel<<<g1(P), 256, 0, stream>>>(
            xsrc, xyzl[i + 1], featl[i], gidx, B, N, M, Cf[i], PING);

        float* src = PING;
        for (int l = 0; l < 3; ++l) {
            float* dst = (src == PING) ? PONG : PING;
            run_layer(sa[i][l], src, dst, P);
            src = dst;
        }
        maxpool_kernel<<<g1((size_t)Cf[i + 1] * B * M), 256, 0, stream>>>(
            src, B, M, Cf[i + 1], featl[i + 1]);
    }

    // =========================== FP modules ===============================
    const float* kfeats = featl[4];
    int Ck = Cf[4]; // 512
    for (int i = 3; i >= 0; --i) {
        const int Nu = Ns[i], Nk = Ns[i + 1];
        const float* unk = (i == 0) ? pc : xyzl[i];
        const float* kn  = xyzl[i + 1];
        const size_t P = (size_t)B * Nu;

        three_nn_kernel<<<g1((size_t)B * Nu), 256, 0, stream>>>(
            unk, kn, B, Nu, Nk, idx3, w3);
        interp_kernel<<<g1((size_t)Ck * P), 256, 0, stream>>>(
            kfeats, idx3, w3, B, Nu, Nk, Ck, PING, P);
        const int Cskip = (i == 0) ? 0 : Cf[i];
        if (Cskip > 0)
            copyrows_kernel<<<g1((size_t)Cskip * P), 256, 0, stream>>>(
                featl[i], PING, Ck, Cskip, P);

        const int nl = FP_NL[i];
        float* src = PING;
        for (int l = 0; l < nl; ++l) {
            float* dst = (l == nl - 1) ? fpout[i]
                                       : ((src == PING) ? PONG : PING);
            run_layer(fp[i][l], src, dst, P);
            src = dst;
        }
        kfeats = fpout[i];
        Ck = fpOutC[i];
    }

    // final: [128][B*8192] -> (B,128,8192)
    (void)out_size;
    final_transpose_kernel<<<g1((size_t)B * 128 * 8192), 256, 0, stream>>>(
        fpout[0], (float*)d_out, B, 128, 8192);
}
